// GATv2_8315056685577
// MI455X (gfx1250) — matrix operand order repaired and checked by reading, not compiled, not run on hardware
//
#include <hip/hip_runtime.h>

#define NN 20000          // nodes
#define NE 160000         // original edges
#define ET (NE + NN)      // edges incl. self loops
#define NH 4              // heads
#define LN_EPS 1e-6f

typedef __attribute__((ext_vector_type(16))) __bf16       v16bf;
typedef __attribute__((ext_vector_type(8)))  float        v8f;
typedef __attribute__((ext_vector_type(4)))  unsigned int u32x4;
union Frag { u32x4 q[2]; v16bf v; };

__device__ __forceinline__ unsigned short f2bf(float f) {
    unsigned int u = __float_as_uint(f);
    u += 0x7FFFu + ((u >> 16) & 1u);          // round-to-nearest-even
    return (unsigned short)(u >> 16);
}
// order-preserving f32 <-> u32 mapping for atomicMax-based segment max
__device__ __forceinline__ unsigned int fenc(float f) {
    unsigned int u = __float_as_uint(f);
    return (u & 0x80000000u) ? ~u : (u | 0x80000000u);
}
__device__ __forceinline__ float fdec(unsigned int u) {
    return __uint_as_float((u & 0x80000000u) ? (u & 0x7FFFFFFFu) : ~u);
}

__global__ void k_fill_u32(unsigned int* __restrict__ p, unsigned int v, int n) {
    int i = blockIdx.x * blockDim.x + threadIdx.x;
    if (i < n) p[i] = v;
}

// W f32 [K_real, NC]  ->  WT bf16 [NC, Kpad] (transposed, zero padded in K)
__global__ void k_wt_bf16(const float* __restrict__ W, unsigned short* __restrict__ WT,
                          int Kreal, int Kpad, int NC) {
    int i = blockIdx.x * blockDim.x + threadIdx.x;
    if (i >= NC * Kpad) return;
    int n = i / Kpad, kp = i % Kpad;
    WT[i] = (kp < Kreal) ? f2bf(W[(size_t)kp * NC + n]) : (unsigned short)0;
}

// x f32 [NN, fin] -> xb bf16 [NN, Kpad]
__global__ void k_x_bf16(const float* __restrict__ X, unsigned short* __restrict__ XB,
                         int fin, int Kpad) {
    int i = blockIdx.x * blockDim.x + threadIdx.x;
    if (i >= NN * Kpad) return;
    int r = i / Kpad, kp = i % Kpad;
    XB[i] = (kp < fin) ? f2bf(X[(size_t)r * fin + kp]) : (unsigned short)0;
}

// D[M,NC] = A_bf16[M,K] @ BT_bf16[NC,K]^T + bias
// One wave computes a 16x64 output tile (4 accumulators, A reused).
// K is a compile-time constant so the whole K-loop fully unrolls: the
// scheduler can hoist load clauses across WMMAs and use partial
// s_wait_loadcnt instead of full drains.
template <int K>
__global__ void k_gemm(const unsigned short* __restrict__ A,
                       const unsigned short* __restrict__ BT,
                       const float* __restrict__ bias,
                       float* __restrict__ D, int NC) {
    int lane = threadIdx.x & 31;
    int l16  = lane & 15, half = lane >> 4;
    int rowA = blockIdx.y * 16 + l16;
    int col0 = blockIdx.x * 64 + l16;         // 4 column tiles, stride 16
    const unsigned short* ar  = A  + (size_t)rowA * K;
    const unsigned short* br0 = BT + (size_t)col0 * K;
    v8f acc[4];
#pragma unroll
    for (int t = 0; t < 4; ++t) acc[t] = (v8f){};
#pragma unroll
    for (int kk = 0; kk < K; kk += 32) {
        Frag a;
        // A 16x32 bf16 layout: lane holds K[half*16 .. +16) of its row
        a.q[0] = *(const u32x4*)(ar + kk + half * 16);
        a.q[1] = *(const u32x4*)(ar + kk + half * 16 + 8);
        const unsigned short* bk = br0 + kk + half * 16;
        Frag b[4];
#pragma unroll
        for (int t = 0; t < 4; ++t) {
            // B 32x16 bf16 layout: lane holds K[half*16 .. +16) of its column
            b[t].q[0] = *(const u32x4*)(bk + (size_t)(16 * t) * K);
            b[t].q[1] = *(const u32x4*)(bk + (size_t)(16 * t) * K + 8);
        }
#pragma unroll
        for (int t = 0; t < 4; ++t)
            acc[t] = __builtin_amdgcn_wmma_f32_16x16x32_bf16(false, a.v, false, b[t].v,
                                                             (short)0, acc[t], false, false);
    }
    int rbase = blockIdx.y * 16 + half * 8;   // C/D: VGPR v -> row v + 8*half
#pragma unroll
    for (int t = 0; t < 4; ++t) {
        int col = col0 + 16 * t;
        float bv = bias[col];
#pragma unroll
        for (int v = 0; v < 8; ++v)
            D[(size_t)(rbase + v) * NC + col] = acc[t][v] + bv;
    }
}

// self-loop 'mean' edge attr: pass 1 scatter sums
__global__ void k_ea_sum(const long long* __restrict__ ei, const float* __restrict__ ea,
                         float* __restrict__ esum, float* __restrict__ ecnt) {
    int e = blockIdx.x * blockDim.x + threadIdx.x;
    if (e >= NE) return;
    int d = (int)ei[NE + e];
    atomicAdd(&ecnt[d], 1.f);
#pragma unroll
    for (int k = 0; k < 8; ++k) atomicAdd(&esum[d * 8 + k], ea[(size_t)e * 8 + k]);
}
__global__ void k_ea_mean(const float* __restrict__ esum, const float* __restrict__ ecnt,
                          float* __restrict__ me) {
    int i = blockIdx.x * blockDim.x + threadIdx.x;
    if (i >= NN * 8) return;
    me[i] = esum[i] / fmaxf(ecnt[i >> 3], 1.f);
}

// alpha[e,h] = sum_c att[h,c]*leakyrelu(xl[s]+xr[d]+eattr@We) ; atomic segment max
__global__ void k_edge_alpha(const long long* __restrict__ ei, const float* __restrict__ eattr,
                             const float* __restrict__ me, const float* __restrict__ xl,
                             const float* __restrict__ xr, const float* __restrict__ We,
                             const float* __restrict__ att, float* __restrict__ alpha,
                             unsigned int* __restrict__ amax, int C) {
    int wid  = (blockIdx.x * blockDim.x + threadIdx.x) >> 5;
    int lane = threadIdx.x & 31;
    int e = wid >> 2, h = wid & 3;
    if (e >= ET) return;
    int HC = NH * C;
    int s, d; const float* ea;
    if (e < NE) { s = (int)ei[e]; d = (int)ei[NE + e]; ea = eattr + (size_t)e * 8; }
    else        { s = d = e - NE; ea = me + (size_t)s * 8; }
    const float* xls = xl + (size_t)s * HC + h * C;
    const float* xrd = xr + (size_t)d * HC + h * C;
    const float* ath = att + h * C;
    __builtin_prefetch(xls + lane, 0, 0);     // global_prefetch_b8: warm gathered rows
    __builtin_prefetch(xrd + lane, 0, 0);
    float e0=ea[0],e1=ea[1],e2=ea[2],e3=ea[3],e4=ea[4],e5=ea[5],e6=ea[6],e7=ea[7];
    float sum = 0.f;
    for (int c = lane; c < C; c += 32) {
        int col = h * C + c;
        float ev = e0*We[col] + e1*We[HC+col] + e2*We[2*HC+col] + e3*We[3*HC+col]
                 + e4*We[4*HC+col] + e5*We[5*HC+col] + e6*We[6*HC+col] + e7*We[7*HC+col];
        float m = xls[c] + xrd[c] + ev;
        m = (m > 0.f) ? m : 0.2f * m;           // leaky relu, slope 0.2
        sum += m * ath[c];
    }
    for (int off = 16; off > 0; off >>= 1) sum += __shfl_xor(sum, off, 32);
    if (lane == 0) {
        alpha[(size_t)e * NH + h] = sum;
        atomicMax(&amax[d * NH + h], fenc(sum));
    }
}

__global__ void k_edge_exp(const long long* __restrict__ ei, float* __restrict__ alpha,
                           const unsigned int* __restrict__ amax, float* __restrict__ denom) {
    int i = blockIdx.x * blockDim.x + threadIdx.x;
    if (i >= ET * NH) return;
    int e = i >> 2, h = i & 3;
    int d = (e < NE) ? (int)ei[NE + e] : (e - NE);
    float ex = __expf(alpha[i] - fdec(amax[d * NH + h]));
    alpha[i] = ex;
    atomicAdd(&denom[d * NH + h], ex);
}

__global__ void k_edge_scatter(const long long* __restrict__ ei, const float* __restrict__ alpha,
                               const float* __restrict__ denom, const float* __restrict__ xl,
                               float* __restrict__ acc, int C) {
    int wid  = (blockIdx.x * blockDim.x + threadIdx.x) >> 5;
    int lane = threadIdx.x & 31;
    int e = wid >> 2, h = wid & 3;
    if (e >= ET) return;
    int HC = NH * C;
    int s, d;
    if (e < NE) { s = (int)ei[e]; d = (int)ei[NE + e]; } else { s = d = e - NE; }
    float w = alpha[(size_t)e * NH + h] / (denom[d * NH + h] + 1e-16f);
    const float* xls = xl + (size_t)s * HC + h * C;
    float* ad = acc + (size_t)d * HC + h * C;
    for (int c = lane; c < C; c += 32)
        atomicAdd(&ad[c], xls[c] * w);
}

// head-mean + bias + (mode: 0=none, 1=elu+LN, 2=elu, 3=sigmoid); wave per node
__global__ void k_finalize(const float* __restrict__ acc, const float* __restrict__ bias,
                           const float* __restrict__ lnw, const float* __restrict__ lnb,
                           float* __restrict__ xout, unsigned short* __restrict__ xb,
                           int kpadNext, int C, int mode) {
    int wid  = (blockIdx.x * blockDim.x + threadIdx.x) >> 5;
    int lane = threadIdx.x & 31;
    if (wid >= NN) return;
    int HC = NH * C;
    const float* row = acc + (size_t)wid * HC;
    float vals[4];
#pragma unroll
    for (int i = 0; i < 4; ++i) vals[i] = 0.f;
    int nch = (C + 31) >> 5;
    for (int i = 0; i < nch; ++i) {
        int c = lane + (i << 5);
        if (c < C) {
            float s = 0.f;
#pragma unroll
            for (int h = 0; h < NH; ++h) s += row[h * C + c];
            float v = s * (1.f / NH) + bias[c];
            if (mode == 1 || mode == 2) v = (v > 0.f) ? v : (__expf(v) - 1.f);
            else if (mode == 3)         v = 1.f / (1.f + __expf(-v));
            vals[i] = v;
        }
    }
    if (mode == 1) {   // layer norm over C
        float s = 0.f;
        for (int i = 0; i < nch; ++i) { int c = lane + (i << 5); if (c < C) s += vals[i]; }
        for (int off = 16; off > 0; off >>= 1) s += __shfl_xor(s, off, 32);
        float mu = s / C, sv = 0.f;
        for (int i = 0; i < nch; ++i) { int c = lane + (i << 5);
            if (c < C) { float dd = vals[i] - mu; sv += dd * dd; } }
        for (int off = 16; off > 0; off >>= 1) sv += __shfl_xor(sv, off, 32);
        float inv = rsqrtf(sv / C + LN_EPS);
        for (int i = 0; i < nch; ++i) { int c = lane + (i << 5);
            if (c < C) vals[i] = (vals[i] - mu) * inv * lnw[c] + lnb[c]; }
    }
    if (xout)
        for (int i = 0; i < nch; ++i) { int c = lane + (i << 5);
            if (c < C) xout[(size_t)wid * C + c] = vals[i]; }
    if (xb) {
        int nkp = kpadNext >> 5;
        for (int i = 0; i < nkp; ++i) {
            int kp = lane + (i << 5);
            xb[(size_t)wid * kpadNext + kp] = (kp < C) ? f2bf(vals[i]) : (unsigned short)0;
        }
    }
}

extern "C" void kernel_launch(void* const* d_in, const int* in_sizes, int n_in,
                              void* d_out, int out_size, void* d_ws, size_t ws_size,
                              hipStream_t stream) {
    (void)in_sizes; (void)n_in; (void)out_size; (void)ws_size;
    const float*     x0    = (const float*)d_in[0];
    const long long* ei    = (const long long*)d_in[1];   // int64 [2, NE]
    const float*     eattr = (const float*)d_in[2];

    char* p = (char*)d_ws;
    auto take = [&](size_t bytes) { void* r = (void*)p; p += (bytes + 255) & ~(size_t)255; return r; };
    float*          xl    = (float*)take((size_t)NN * 512 * 4);
    float*          xr    = (float*)take((size_t)NN * 512 * 4);
    float*          acc   = (float*)take((size_t)NN * 512 * 4);
    unsigned short* xb    = (unsigned short*)take((size_t)NN * 128 * 2);
    unsigned short* wlt   = (unsigned short*)take((size_t)512 * 128 * 2);
    unsigned short* wrt   = (unsigned short*)take((size_t)512 * 128 * 2);
    float*          alpha = (float*)take((size_t)ET * NH * 4);
    unsigned int*   amax  = (unsigned int*)take((size_t)NN * NH * 4);
    float*          denom = (float*)take((size_t)NN * NH * 4);
    float*          esum  = (float*)take((size_t)NN * 8 * 4);
    float*          ecnt  = (float*)take((size_t)NN * 4);
    float*          me    = (float*)take((size_t)NN * 8 * 4);

    // output[0] = node_features passthrough
    hipMemcpyAsync(d_out, (const void*)x0, (size_t)NN * 16 * 4, hipMemcpyDeviceToDevice, stream);

    // self-loop mean edge attrs
    k_fill_u32<<<(NN * 8 + 255) / 256, 256, 0, stream>>>((unsigned int*)esum, 0u, NN * 8);
    k_fill_u32<<<(NN + 255) / 256, 256, 0, stream>>>((unsigned int*)ecnt, 0u, NN);
    k_ea_sum<<<(NE + 255) / 256, 256, 0, stream>>>(ei, eattr, esum, ecnt);
    k_ea_mean<<<(NN * 8 + 255) / 256, 256, 0, stream>>>(esum, ecnt, me);

    auto launch_gemm = [&](const unsigned short* A, const unsigned short* BT,
                           const float* bias, float* D, int K, int NC) {
        dim3 gg(NC / 64, NN / 16);            // 16x64 tile per wave
        if (K == 32)       k_gemm<32><<<gg, 32, 0, stream>>>(A, BT, bias, D, NC);
        else if (K == 64)  k_gemm<64><<<gg, 32, 0, stream>>>(A, BT, bias, D, NC);
        else               k_gemm<128><<<gg, 32, 0, stream>>>(A, BT, bias, D, NC);
    };

    struct LayerCfg { int base; int fin; int C; int mode; int lnIdx; };
    const LayerCfg L[6] = {
        {3, 16, 128, 1, 45}, {10, 128, 128, 1, 47}, {17, 128, 128, 1, 49},
        {24, 128, 64, 0, -1}, {31, 64, 32, 2, -1}, {38, 32, 16, 3, -1},
    };

    for (int li = 0; li < 6; ++li) {
        const LayerCfg& c = L[li];
        const float* Wl   = (const float*)d_in[c.base + 0];
        const float* bl   = (const float*)d_in[c.base + 1];
        const float* Wr   = (const float*)d_in[c.base + 2];
        const float* br   = (const float*)d_in[c.base + 3];
        const float* We   = (const float*)d_in[c.base + 4];
        const float* att  = (const float*)d_in[c.base + 5];
        const float* bias = (const float*)d_in[c.base + 6];
        int HC = NH * c.C;
        int K  = (c.fin < 32) ? 32 : c.fin;   // pad K for WMMA 16x16x32

        k_wt_bf16<<<(HC * K + 255) / 256, 256, 0, stream>>>(Wl, wlt, c.fin, K, HC);
        k_wt_bf16<<<(HC * K + 255) / 256, 256, 0, stream>>>(Wr, wrt, c.fin, K, HC);
        if (li == 0)
            k_x_bf16<<<(NN * K + 255) / 256, 256, 0, stream>>>(x0, xb, c.fin, K);

        launch_gemm(xb, wlt, bl, xl, K, HC);
        launch_gemm(xb, wrt, br, xr, K, HC);

        k_fill_u32<<<(NN * NH + 255) / 256, 256, 0, stream>>>(amax, 0x007FFFFFu, NN * NH); // enc(-inf)
        k_fill_u32<<<(NN * NH + 255) / 256, 256, 0, stream>>>((unsigned int*)denom, 0u, NN * NH);
        k_fill_u32<<<(NN * HC + 255) / 256, 256, 0, stream>>>((unsigned int*)acc, 0u, NN * HC);

        int nwaves = ET * NH;
        k_edge_alpha<<<(nwaves + 3) / 4, 128, 0, stream>>>(ei, eattr, me, xl, xr, We, att, alpha, amax, c.C);
        k_edge_exp<<<(ET * NH + 255) / 256, 256, 0, stream>>>(ei, alpha, amax, denom);
        k_edge_scatter<<<(nwaves + 3) / 4, 128, 0, stream>>>(ei, alpha, denom, xl, acc, c.C);

        float* xout = nullptr;
        unsigned short* xbn = nullptr;
        int kpn = 0;
        const float *lnw = nullptr, *lnb = nullptr;
        if (c.lnIdx >= 0) { lnw = (const float*)d_in[c.lnIdx]; lnb = (const float*)d_in[c.lnIdx + 1]; }
        if (li == 5) xout = (float*)d_out + (size_t)NN * 16;   // output[1] = r
        else { xbn = xb; kpn = (L[li + 1].fin < 32) ? 32 : L[li + 1].fin; }
        k_finalize<<<(NN + 3) / 4, 128, 0, stream>>>(acc, bias, lnw, lnb, xout, xbn, kpn, c.C, c.mode);
    }
}
